// GPNN_HICO_74680891343711
// MI455X (gfx1250) — matrix operand order repaired and checked by reading, not compiled, not run on hardware
//
#include <hip/hip_runtime.h>
#include <hip/hip_bf16.h>
#include <math.h>

// ---------------------------------------------------------------------------
// Problem constants
// ---------------------------------------------------------------------------
#define BB   16
#define NN   64
#define FE   1216
#define FN   1108
#define MD   256
#define HD   256
#define KC   600
#define NVW  4096           // N*N rows per batch (flattened v*64+w)

typedef __attribute__((ext_vector_type(16))) _Float16 v16h;
typedef __attribute__((ext_vector_type(8)))  float    v8f;

// ---------------------------------------------------------------------------
// WMMA fragment helpers (layouts per CDNA5 ISA 7.12.2, wave32)
// A (16x32 f16): lane L holds row M=L%16; VGPR0-3 hold K = 8*(L/16)+0..7,
//                VGPR4-7 hold K = 16 + 8*(L/16)+0..7.
// B (32x16 f16): lane L holds col N=L%16; VGPR0-3 hold K = 8*(L/16)+0..7,
//                VGPR4-7 hold K = 16 + 8*(L/16)+0..7 (same as A).
// C (16x16 f32): VGPR v at lane L is element [row = v + 8*(L/16)][col = L%16].
// ---------------------------------------------------------------------------
__device__ __forceinline__ v16h frag_a_from_f16(const _Float16* base, int stride) {
  int lane = threadIdx.x & 31;
  int row  = lane & 15;
  int kb   = (lane >> 4) * 8;
  const _Float16* p0 = base + row * stride + kb;
  const _Float16* p1 = p0 + 16;
  v16h a;
#pragma unroll
  for (int h = 0; h < 8; ++h) { a[h] = p0[h]; a[h + 8] = p1[h]; }
  return a;
}

__device__ __forceinline__ v16h frag_a_from_f32(const float* base, int stride) {
  int lane = threadIdx.x & 31;
  int row  = lane & 15;
  int kb   = (lane >> 4) * 8;
  const float* p0 = base + (long)row * stride + kb;
  const float* p1 = p0 + 16;
  v16h a;
#pragma unroll
  for (int h = 0; h < 8; ++h) {
    a[h]     = (_Float16)p0[h];
    a[h + 8] = (_Float16)p1[h];
  }
  return a;
}

// W is row-major (Nrows, Kcols); B[k][n] = W[n0+n][k0+k]; base = W + n0*stride + k0
__device__ __forceinline__ v16h frag_b_from_f16(const _Float16* base, int stride) {
  int lane = threadIdx.x & 31;
  int n    = lane & 15;
  int kb   = (lane >> 4) * 8;
  const _Float16* p = base + n * stride + kb;
  v16h b;
#pragma unroll
  for (int h = 0; h < 8; ++h) { b[h] = p[h]; b[h + 8] = p[h + 16]; }
  return b;
}

__device__ __forceinline__ float sigm(float x) { return 1.0f / (1.0f + __expf(-x)); }

// CDNA5 async global->LDS copy (ASYNCcnt-tracked, CDNA5 ISA).
// VDST VGPR holds the per-lane LDS byte address (generic LDS pointers carry the
// group-segment offset in their low 32 bits), VADDR is the 64-bit global addr.
__device__ __forceinline__ void async_cp_b128(const void* gptr, void* lptr) {
  unsigned int lds = (unsigned int)(unsigned long long)lptr;
  asm volatile("global_load_async_to_lds_b128 %0, %1, off"
               :: "v"(lds), "v"(gptr) : "memory");
}
__device__ __forceinline__ void wait_asynccnt0() {
  asm volatile("s_wait_asynccnt 0x0" ::: "memory");
}

// ---------------------------------------------------------------------------
// K_conv: convert weight matrices to f16 in workspace
//   region0: W_er (256x1216), region1: W_ev = Wm[:,256:512], region2: Wl1
// ---------------------------------------------------------------------------
__global__ __launch_bounds__(256) void k_conv(const float* W_er, const float* Wm,
                                              const float* Wl1, _Float16* Werh,
                                              _Float16* Wevh, _Float16* Wl1h) {
  long i = (long)blockIdx.x * 256 + threadIdx.x;
  const long n0 = (long)MD * FE;          // 311296
  if (i < n0) {
    Werh[i] = (_Float16)W_er[i];
  } else if (i < n0 + MD * MD) {
    long j = i - n0;
    int c = (int)(j >> 8), m = (int)(j & 255);
    Wevh[j] = (_Float16)Wm[(long)c * (2 * MD) + MD + m];
  } else {
    long j = i - n0 - MD * MD;
    Wl1h[j] = (_Float16)Wl1[j];
  }
}

// ---------------------------------------------------------------------------
// K_xp: Xp[b][m][v] = node_features[b,v,:] . W_nr[m,:] + b_nr[m]
//   grid = (b*64+v), block = 256 (m)
// ---------------------------------------------------------------------------
__global__ __launch_bounds__(256) void k_xp(const float* node, const float* Wnr,
                                            const float* bnr, float* Xp) {
  __shared__ float sn[FN];
  int b = blockIdx.x >> 6, v = blockIdx.x & 63;
  const float* nf = node + ((long)b * NN + v) * FN;
  for (int q = threadIdx.x; q < FN; q += 256) sn[q] = nf[q];
  __syncthreads();
  int m = threadIdx.x;
  const float* wr = Wnr + (long)m * FN;
  float acc = bnr[m];
  for (int f = 0; f < FN; ++f) acc = fmaf(sn[f], wr[f], acc);
  Xp[((long)b * MD + m) * NN + v] = acc;
}

// ---------------------------------------------------------------------------
// K_hw: Hw[b][c][w] = sum_m Wm[c][m] * Xp[b][m][w]   (W_hw = Wm[:, :256])
//   grid = (b*64+w), block = 256 (c)
// ---------------------------------------------------------------------------
__global__ __launch_bounds__(256) void k_hw(const float* Xp, const float* Wm,
                                            float* Hw) {
  __shared__ float sx[MD];
  int b = blockIdx.x >> 6, w = blockIdx.x & 63;
  sx[threadIdx.x] = Xp[((long)b * MD + threadIdx.x) * NN + w];
  __syncthreads();
  int c = threadIdx.x;
  const float* wr = Wm + (long)c * (2 * MD);
  float acc = 0.0f;
  for (int m = 0; m < MD; ++m) acc = fmaf(wr[m], sx[m], acc);
  Hw[((long)b * MD + c) * NN + w] = acc;
}

// ---------------------------------------------------------------------------
// K1: fused  E0 = edge @ W_er^T + b_er   (WMMA, K=1216, async-staged B)
//            Mpre = relu(Hw + E0 @ W_ev^T + bm)  -> f16 scratch [b][vw][c]
//            h1 = relu(E0 @ Wl1^T + bl1); A1 = Wl2.h1 + bl2; S1 = sigmoid(A1)
//   grid = b*64 + row-tile(64 rows), block = 256 (8 waves)
//   wave w: row-group rg = w>>1 (16 rows), col-half ch = w&1 (128 cols)
// ---------------------------------------------------------------------------
__global__ __launch_bounds__(256) void k1(const float* edge, const _Float16* Werh,
                                          const float* b_er, const _Float16* Wevh,
                                          const float* Hw, const float* bm,
                                          const _Float16* Wl1h, const float* bl1,
                                          const float* Wl2, const float* bl2,
                                          _Float16* MpreH, float* S1) {
  __shared__ _Float16 sB[2][32 * MD];     // double-buffered B tile [m][k], 2x16KB
  __shared__ _Float16 sE0[64 * MD];       // E0 tile, 32KB
  __shared__ float sAred[64];
  int b  = blockIdx.x >> 6;
  int r0 = (blockIdx.x & 63) << 6;
  int tid  = threadIdx.x;
  int wave = tid >> 5;
  int lane = tid & 31;
  int rg   = wave >> 1;                   // 0..3
  int ch   = wave & 1;                    // 0..1
  int col  = lane & 15;
  int rb   = (lane >> 4) * 8;

  // ---- stage 1: E0 rows [r0, r0+64) x all 256 cols, K over 1216 ----
  // async-stage B tile (Werh rows 0..255, k0..k0+31) into LDS: thread t = row m
  const _Float16* gB = Werh + (long)tid * FE;
  const float* Arows = edge + ((long)b * NVW + r0 + rg * 16) * FE;

  {
    _Float16* l = &sB[0][tid * 32];
#pragma unroll
    for (int q = 0; q < 4; ++q) async_cp_b128(gB + q * 8, l + q * 8);
  }
  wait_asynccnt0();
  __syncthreads();

  v8f acc[8] = {};
  const int NS = FE / 32;                 // 38
  for (int s = 0; s < NS; ++s) {
    int k0 = s * 32;
    if (s + 1 < NS) {                     // prefetch next B tile
      _Float16* l = &sB[(s + 1) & 1][tid * 32];
      const _Float16* g = gB + k0 + 32;
#pragma unroll
      for (int q = 0; q < 4; ++q) async_cp_b128(g + q * 8, l + q * 8);
    }
    v16h a = frag_a_from_f32(Arows + k0, FE);
    const _Float16* Bb = &sB[s & 1][0];
#pragma unroll
    for (int t = 0; t < 8; ++t) {
      v16h bf = frag_b_from_f16(Bb + (ch * 128 + t * 16) * 32, 32);
      acc[t] = __builtin_amdgcn_wmma_f32_16x16x32_f16(false, a, false, bf,
                                                      (short)0, acc[t], false, false);
    }
    wait_asynccnt0();                     // next buffer resident ...
    __syncthreads();                      // ... and this buffer fully consumed
  }

  // add b_er, park E0 tile in LDS as f16 (A operand for stage-2 GEMMs)
#pragma unroll
  for (int t = 0; t < 8; ++t) {
    int m = ch * 128 + t * 16 + col;
    float bias = b_er[m];
#pragma unroll
    for (int v = 0; v < 8; ++v)
      sE0[(rg * 16 + rb + v) * MD + m] = (_Float16)(acc[t][v] + bias);
  }
  if (tid < 64) sAred[tid] = 0.0f;
  __syncthreads();

  // ---- stage 2a: Mpre = relu(Hw + E0 @ W_ev^T + bm), store f16 ----
  v8f mc[8] = {};
  for (int k0 = 0; k0 < MD; k0 += 32) {
    v16h a = frag_a_from_f16(sE0 + (rg * 16) * MD + k0, MD);
#pragma unroll
    for (int t = 0; t < 8; ++t) {
      v16h bf = frag_b_from_f16(Wevh + (long)(ch * 128 + t * 16) * MD + k0, MD);
      mc[t] = __builtin_amdgcn_wmma_f32_16x16x32_f16(false, a, false, bf,
                                                     (short)0, mc[t], false, false);
    }
  }
#pragma unroll
  for (int t = 0; t < 8; ++t) {
    int c = ch * 128 + t * 16 + col;
    float bmc = bm[c];
#pragma unroll
    for (int v = 0; v < 8; ++v) {
      int r = r0 + rg * 16 + rb + v;               // r = i*64 + w
      float val = mc[t][v] + Hw[((long)b * MD + c) * NN + (r & 63)] + bmc;
      val = fmaxf(val, 0.0f);
      MpreH[((long)b * NVW + r) * MD + c] = (_Float16)val;
    }
  }

  // ---- stage 2b: h1 = relu(E0 @ Wl1^T + bl1); A1 = Wl2.h1 + bl2 ----
  v8f hc[8] = {};
  for (int k0 = 0; k0 < MD; k0 += 32) {
    v16h a = frag_a_from_f16(sE0 + (rg * 16) * MD + k0, MD);
#pragma unroll
    for (int t = 0; t < 8; ++t) {
      v16h bf = frag_b_from_f16(Wl1h + (long)(ch * 128 + t * 16) * MD + k0, MD);
      hc[t] = __builtin_amdgcn_wmma_f32_16x16x32_f16(false, a, false, bf,
                                                     (short)0, hc[t], false, false);
    }
  }
#pragma unroll
  for (int t = 0; t < 8; ++t) {
    int hcol = ch * 128 + t * 16 + col;
    float wl2 = Wl2[hcol];
    float bl  = bl1[hcol];
#pragma unroll
    for (int v = 0; v < 8; ++v) {
      float h1v = fmaxf(hc[t][v] + bl, 0.0f);
      atomicAdd(&sAred[rg * 16 + rb + v], h1v * wl2);
    }
  }
  __syncthreads();
  if (tid < 64) {
    float A1 = sAred[tid] + bl2[0];
    S1[(long)b * NVW + r0 + tid] = sigm(A1);
  }
}

// ---------------------------------------------------------------------------
// K2: iteration 2.  E2 row (v,w) = Mpre row (w*64+v) * S1[w*64+v]  (transpose
//     is a pure row permutation given Mpre layout [b][vw][c]).
//     h1' = relu(E2 @ Wl1^T + bl1); A2 = Wl2.h1' + bl2  -> pred_adj_mat, S2.
//   grid = b*256 + row-tile, block = 128
// ---------------------------------------------------------------------------
__global__ __launch_bounds__(128) void k2(const _Float16* MpreH, const float* S1,
                                          const _Float16* Wl1h, const float* bl1,
                                          const float* Wl2, const float* bl2,
                                          float* adj_out, float* S2) {
  __shared__ _Float16 sE[16 * MD];
  __shared__ float sA[16];
  int b  = blockIdx.x >> 8;
  int r0 = (blockIdx.x & 255) << 4;
  int v0 = r0 >> 6;
  int w0 = r0 & 63;

  // stage the permuted+scaled tile: 16 rows x 256 f16
  {
    int i   = threadIdx.x >> 3;        // row 0..15
    int seg = threadIdx.x & 7;         // 32 halves each
    int srcr = (w0 + i) * NN + v0;
    float s = S1[(long)b * NVW + srcr];
    const _Float16* src = MpreH + ((long)b * NVW + srcr) * MD + seg * 32;
    _Float16* dst = sE + i * MD + seg * 32;
#pragma unroll
    for (int q = 0; q < 32; ++q) dst[q] = (_Float16)((float)src[q] * s);
  }
  if (threadIdx.x < 16) sA[threadIdx.x] = 0.0f;
  __syncthreads();

  int wave = threadIdx.x >> 5;
  int lane = threadIdx.x & 31;
  int col  = lane & 15;
  int rb   = (lane >> 4) * 8;

  v8f hc[4] = {};
  for (int k0 = 0; k0 < MD; k0 += 32) {
    v16h a = frag_a_from_f16(sE + k0, MD);
#pragma unroll
    for (int t = 0; t < 4; ++t) {
      v16h bf = frag_b_from_f16(Wl1h + (long)(wave * 64 + t * 16) * MD + k0, MD);
      hc[t] = __builtin_amdgcn_wmma_f32_16x16x32_f16(false, a, false, bf,
                                                     (short)0, hc[t], false, false);
    }
  }
#pragma unroll
  for (int t = 0; t < 4; ++t) {
    int hcol = wave * 64 + t * 16 + col;
    float wl2 = Wl2[hcol];
    float bl  = bl1[hcol];
#pragma unroll
    for (int v = 0; v < 8; ++v) {
      float h1v = fmaxf(hc[t][v] + bl, 0.0f);
      atomicAdd(&sA[rb + v], h1v * wl2);
    }
  }
  __syncthreads();
  if (threadIdx.x < 16) {
    int r = r0 + threadIdx.x;
    float A2 = sA[threadIdx.x] + bl2[0];
    adj_out[(long)b * NVW + r] = A2;          // pred_adj_mat (pre-sigmoid)
    S2[(long)b * NVW + r] = sigm(A2);
  }
}

// ---------------------------------------------------------------------------
// K3: m_sum[b,v,c] = sum_w Mpre[b][v*64+w][c] * S2[b][v*64+w]
//     then GRU cell + label readout per (b,v).   grid = b*64+v, block = 256
// ---------------------------------------------------------------------------
__global__ __launch_bounds__(256) void k3(const _Float16* MpreH, const float* S2,
                                          const float* Xp, const float* Wi,
                                          const float* bi, const float* Wh,
                                          const float* bh, const float* Wro,
                                          const float* bro, float* lab_out) {
  __shared__ float s_s2[NN];
  __shared__ float s_x[MD];
  __shared__ float s_h[MD];
  __shared__ float s_hn[MD];
  int b = blockIdx.x >> 6, v = blockIdx.x & 63;
  int t = threadIdx.x;
  if (t < NN) s_s2[t] = S2[(long)b * NVW + v * NN + t];
  __syncthreads();

  float msum = 0.0f;
  const _Float16* mrow = MpreH + ((long)b * NVW + (long)v * NN) * MD + t;
  for (int w = 0; w < NN; ++w) msum = fmaf((float)mrow[(long)w * MD], s_s2[w], msum);
  float hv = Xp[((long)b * MD + t) * NN + v];
  s_x[t] = msum;
  s_h[t] = hv;
  __syncthreads();

  float ir = bi[t], iz = bi[t + MD], in_ = bi[t + 2 * MD];
  float hr = bh[t], hz = bh[t + MD], hn_ = bh[t + 2 * MD];
  const float* wi0 = Wi + (long)t * MD;
  const float* wi1 = Wi + (long)(t + MD) * MD;
  const float* wi2 = Wi + (long)(t + 2 * MD) * MD;
  const float* wh0 = Wh + (long)t * MD;
  const float* wh1 = Wh + (long)(t + MD) * MD;
  const float* wh2 = Wh + (long)(t + 2 * MD) * MD;
  for (int c = 0; c < MD; ++c) {
    float xv = s_x[c], hc = s_h[c];
    ir  = fmaf(xv, wi0[c], ir);
    iz  = fmaf(xv, wi1[c], iz);
    in_ = fmaf(xv, wi2[c], in_);
    hr  = fmaf(hc, wh0[c], hr);
    hz  = fmaf(hc, wh1[c], hz);
    hn_ = fmaf(hc, wh2[c], hn_);
  }
  float r = sigm(ir + hr);
  float z = sigm(iz + hz);
  float n = tanhf(in_ + r * hn_);
  s_hn[t] = (1.0f - z) * n + z * hv;
  __syncthreads();

  for (int j = t; j < KC; j += 256) {
    const float* wr = Wro + (long)j * MD;
    float acc = bro[j];
    for (int c = 0; c < MD; ++c) acc = fmaf(s_hn[c], wr[c], acc);
    lab_out[((long)b * NN + v) * KC + j] = acc;
  }
}

// ---------------------------------------------------------------------------
// kernel_launch
// ---------------------------------------------------------------------------
extern "C" void kernel_launch(void* const* d_in, const int* in_sizes, int n_in,
                              void* d_out, int out_size, void* d_ws, size_t ws_size,
                              hipStream_t stream) {
  (void)in_sizes; (void)n_in; (void)out_size; (void)ws_size;
  const float* edge = (const float*)d_in[0];
  const float* node = (const float*)d_in[1];
  const float* W_er = (const float*)d_in[7];
  const float* b_er = (const float*)d_in[8];
  const float* W_nr = (const float*)d_in[9];
  const float* b_nr = (const float*)d_in[10];
  const float* Wl1  = (const float*)d_in[11];
  const float* bl1  = (const float*)d_in[12];
  const float* Wl2  = (const float*)d_in[13];
  const float* bl2  = (const float*)d_in[14];
  const float* Wm   = (const float*)d_in[15];
  const float* bm   = (const float*)d_in[16];
  const float* Wi   = (const float*)d_in[17];
  const float* bi   = (const float*)d_in[18];
  const float* Wh   = (const float*)d_in[19];
  const float* bh   = (const float*)d_in[20];
  const float* Wro  = (const float*)d_in[21];
  const float* bro  = (const float*)d_in[22];

  char* ws = (char*)d_ws;
  _Float16* Werh  = (_Float16*)(ws + 0);          // 256*1216*2 = 622592
  _Float16* Wevh  = (_Float16*)(ws + 622592);     // 256*256*2  = 131072
  _Float16* Wl1h  = (_Float16*)(ws + 753664);     // 131072
  float*    Xp    = (float*)(ws + 884736);        // 16*256*64*4 = 1048576
  float*    Hw    = (float*)(ws + 1933312);       // 1048576
  float*    S1    = (float*)(ws + 2981888);       // 16*4096*4 = 262144
  float*    S2    = (float*)(ws + 3244032);       // 262144
  _Float16* MpreH = (_Float16*)(ws + 3506176);    // 16*4096*256*2 = 33554432

  float* adj_out = (float*)d_out;                 // (16,64,64)
  float* lab_out = adj_out + BB * NVW;            // (16,64,600)

  k_conv<<<1728, 256, 0, stream>>>(W_er, Wm, Wl1, Werh, Wevh, Wl1h);
  k_xp  <<<BB * NN, 256, 0, stream>>>(node, W_nr, b_nr, Xp);
  k_hw  <<<BB * NN, 256, 0, stream>>>(Xp, Wm, Hw);
  k1    <<<BB * NN, 256, 0, stream>>>(edge, Werh, b_er, Wevh, Hw, bm,
                                      Wl1h, bl1, Wl2, bl2, MpreH, S1);
  k2    <<<BB * 256, 128, 0, stream>>>(MpreH, S1, Wl1h, bl1, Wl2, bl2,
                                       adj_out, S2);
  k3    <<<BB * NN, 256, 0, stream>>>(MpreH, S2, Xp, Wi, bi, Wh, bh,
                                      Wro, bro, lab_out);
}
